// ConvolutionalAttention_36043365548745
// MI455X (gfx1250) — matrix operand order repaired and checked by reading, not compiled, not run on hardware
//
#include <hip/hip_runtime.h>
#include <hip/hip_bf16.h>
#include <stdint.h>

// ---------------------------------------------------------------------------
// CDNA5 (gfx1250) bf16 WMMA building blocks
// ---------------------------------------------------------------------------
typedef __attribute__((ext_vector_type(16))) __bf16 v16bf;
typedef __attribute__((ext_vector_type(8)))  float  v8f;
typedef __attribute__((ext_vector_type(4)))  unsigned int u32x4;
typedef __attribute__((ext_vector_type(8)))  int i32x8;
typedef __attribute__((ext_vector_type(4)))  int i32x4;

#if defined(__gfx1250__) && __has_builtin(__builtin_amdgcn_tensor_load_to_lds)
#define HAVE_TDM 1
#else
#define HAVE_TDM 0
#endif

__device__ __forceinline__ unsigned short f2bf(float f) {
  union { float f; unsigned u; } x; x.f = f;
  unsigned r = x.u + 0x7FFFu + ((x.u >> 16) & 1u);   // round-to-nearest-even
  return (unsigned short)(r >> 16);
}

union FragBits {
  v16bf v;
  struct { uint4 lo, hi; } q;
};

// A-matrix fragment (16 x 32 bf16 tile, row-major, `ld` elements per row).
// ISA layout: lanes 0-15 -> rows 0-15, K in {0..7, 16..23};
//             lanes 16-31 -> rows 0-15, K in {8..15, 24..31}.
__device__ __forceinline__ v16bf frag_a(const unsigned short* tile, int ld) {
  int lane = threadIdx.x & 31;
  int row  = lane & 15;
  int kb   = (lane >> 4) << 3;                // 0 or 8
  const unsigned short* p = tile + row * ld + kb;
  FragBits f;
  f.q.lo = *(const uint4*)(p);                // K = kb .. kb+7
  f.q.hi = *(const uint4*)(p + 16);           // K = kb+16 .. kb+23
  return f.v;
}

// B-matrix fragment (32 x 16 bf16 K x N tile stored row-major [K][ld]).
// Lane l holds column N=l%16, K in {kb..kb+7, kb+16..kb+23}, kb = 8*(l/16).
__device__ __forceinline__ v16bf frag_b(const unsigned short* tile, int ld) {
  int lane = threadIdx.x & 31;
  int kb   = (lane >> 4) << 3;
  const unsigned short* p = tile + (lane & 15);
  unsigned short t[16];
#pragma unroll
  for (int s = 0; s < 16; ++s) {
    int k = (s < 8) ? (kb + s) : (16 + kb + (s - 8));
    t[s] = p[k * ld];
  }
  FragBits f;
  __builtin_memcpy(&f, t, sizeof(t));
  return f.v;
}

__device__ __forceinline__ v8f wmma_bf16(v16bf a, v16bf b, v8f c) {
  // D = A(16x32) * B(32x16) + C, f32 accumulate
  return __builtin_amdgcn_wmma_f32_16x16x32_bf16(
      /*neg_a=*/false, a, /*neg_b=*/false, b,
      /*c_mod=*/(short)0, c, /*reuse_a=*/false, /*reuse_b=*/false);
}

// ---------------------------------------------------------------------------
// Elementwise / prep kernels
// ---------------------------------------------------------------------------
__global__ void k_cvt_bf16(const float* __restrict__ x,
                           unsigned short* __restrict__ y, size_t n) {
  size_t i = (size_t)blockIdx.x * blockDim.x + threadIdx.x;
  if (i < n) y[i] = f2bf(x[i]);
}

// HWIO f32 weights -> [Co][9*Ci] bf16 (K index = (ky*3+kx)*Ci + ci)
__global__ void k_prep_w(const float* __restrict__ w,
                         unsigned short* __restrict__ o, int Ci, int Co) {
  int e = blockIdx.x * blockDim.x + threadIdx.x;
  int K9 = 9 * Ci;
  if (e >= K9 * Co) return;
  int co = e / K9, k = e % K9;
  o[e] = f2bf(w[(size_t)k * Co + co]);
}

// ---------------------------------------------------------------------------
// Implicit-GEMM 3x3 SAME conv, bf16 WMMA, f32 out.  Cout fixed at 64.
// block = 128 threads (4 waves); block computes 64 pixels x 64 out-channels
// (4 accumulators / wave -> 4 WMMAs per staged A tile).  grid = M/64.
// Tap loop is outermost (unrolled: dy/dx/bounds hoisted to constants).
// Cin must be a multiple of 32 (64 or 512 here).
// ---------------------------------------------------------------------------
__global__ __launch_bounds__(128)
void k_conv3x3_wmma(const unsigned short* __restrict__ xbf,   // [B*H*W][Cin]
                    const unsigned short* __restrict__ wbf,   // [64][9*Cin]
                    const float* __restrict__ bias,           // [64]
                    float* __restrict__ out,                  // [B*H*W][64]
                    int H, int W, int Cin, int relu) {
  __shared__ __align__(16) unsigned short At[64 * 32];   // [pix][k]
  __shared__ __align__(16) unsigned short Bt[32 * 64];   // [k][co]
  const int tid = threadIdx.x;
  const int lane = tid & 31, wave = tid >> 5;
  const int m0 = blockIdx.x * 64;
  const int K9 = 9 * Cin;

  // staging role: 64 rows x 2 halves of 16 channels
  const int arow = tid >> 1, ahalf = tid & 1;
  const int am = m0 + arow;
  const int ax = am % W;
  const int ay = (am / W) % H;
  const int ab = am / (W * H);

  const v8f vzero = {0.f, 0.f, 0.f, 0.f, 0.f, 0.f, 0.f, 0.f};
  v8f acc[4];
#pragma unroll
  for (int nt = 0; nt < 4; ++nt) acc[nt] = vzero;

#pragma unroll
  for (int tap = 0; tap < 9; ++tap) {
    const int dy = tap / 3 - 1, dx = tap % 3 - 1;   // constants after unroll
    const int yy = ay + dy, xx = ax + dx;
    const bool ok = (yy >= 0) && (yy < H) && (xx >= 0) && (xx < W);
    const unsigned short* srcp =
        xbf + (((size_t)ab * H + yy) * W + xx) * Cin + ahalf * 16;
    const int kbase = tap * Cin;
    for (int kc = 0; kc < Cin; kc += 32) {
      uint4 z0 = make_uint4(0u, 0u, 0u, 0u), z1 = make_uint4(0u, 0u, 0u, 0u);
      if (ok) {
        const uint4* s = (const uint4*)(srcp + kc);
        z0 = s[0]; z1 = s[1];
      }
      *(uint4*)(At + arow * 32 + ahalf * 16)     = z0;
      *(uint4*)(At + arow * 32 + ahalf * 16 + 8) = z1;
      const int k0 = kbase + kc;
      for (int e = tid; e < 2048; e += 128) {       // B tile: 32 k x 64 co
        int k = e >> 6, n = e & 63;
        Bt[e] = wbf[(size_t)n * K9 + k0 + k];
      }
      __syncthreads();
      v16bf a = frag_a(At + wave * 16 * 32, 32);
#pragma unroll
      for (int nt = 0; nt < 4; ++nt) {
        v16bf b = frag_b(Bt + nt * 16, 64);
        acc[nt] = wmma_bf16(a, b, acc[nt]);
      }
      __syncthreads();
    }
  }

  // C/D layout: VGPR r, lane<16 -> row r; lane>=16 -> row r+8; col = lane%16
  const int col = lane & 15, radd = (lane >> 4) << 3;
#pragma unroll
  for (int nt = 0; nt < 4; ++nt) {
    const int co = nt * 16 + col;
    const float bv = bias[co];
#pragma unroll
    for (int r = 0; r < 8; ++r) {
      float v = acc[nt][r] + bv;
      if (relu) v = fmaxf(v, 0.f);
      int m = m0 + wave * 16 + r + radd;
      out[(size_t)m * 64 + co] = v;
    }
  }
}

// ---------------------------------------------------------------------------
// BatchNorm (training mode): stats, finalize, apply
// ---------------------------------------------------------------------------
__global__ void k_bn_stats(const float* __restrict__ x, int M,
                           float* __restrict__ gsum, float* __restrict__ gsq) {
  __shared__ float ssum[256];
  __shared__ float ssq[256];
  const int tid = threadIdx.x;
  const int c = tid & 63, sub = tid >> 6;          // 4 row-groups per block
  float s = 0.f, q = 0.f;
  for (int m = blockIdx.x * 4 + sub; m < M; m += gridDim.x * 4) {
    float v = x[(size_t)m * 64 + c];
    s += v; q += v * v;
  }
  ssum[tid] = s; ssq[tid] = q;
  __syncthreads();
  if (tid < 64) {
    s = ssum[tid] + ssum[tid + 64] + ssum[tid + 128] + ssum[tid + 192];
    q = ssq[tid]  + ssq[tid + 64]  + ssq[tid + 128]  + ssq[tid + 192];
    atomicAdd(&gsum[tid], s);
    atomicAdd(&gsq[tid], q);
  }
}

__global__ void k_bn_finalize(const float* __restrict__ gsum,
                              const float* __restrict__ gsq,
                              const float* __restrict__ gamma,
                              const float* __restrict__ beta,
                              float Mf, float* __restrict__ scale,
                              float* __restrict__ shift) {
  int c = threadIdx.x;
  float mean = gsum[c] / Mf;
  float var  = gsq[c] / Mf - mean * mean;
  float s = gamma[c] * rsqrtf(var + 1e-3f);
  scale[c] = s;
  shift[c] = beta[c] - mean * s;
}

// BN apply + write BOTH attention layouts:
// T1[((b*64+c)*256 + i)*256 + j]  (time-major)
// T2[((b*64+c)*256 + j)*256 + i]  (freq-major)
__global__ void k_bn_apply_tr(const float* __restrict__ x,
                              const float* __restrict__ sc,
                              const float* __restrict__ sh,
                              unsigned short* __restrict__ T1,
                              unsigned short* __restrict__ T2) {
  size_t e = (size_t)blockIdx.x * 256 + threadIdx.x;
  int c = (int)(e & 63);
  size_t pix = e >> 6;
  int j = (int)(pix & 255);
  int i = (int)((pix >> 8) & 255);
  int b = (int)(pix >> 16);
  unsigned short v = f2bf(x[e] * sc[c] + sh[c]);
  size_t base = ((size_t)b * 64 + c) * 256;
  T1[(base + i) * 256 + j] = v;
  T2[(base + j) * 256 + i] = v;
}

__global__ void k_bn_apply_bf16(const float* __restrict__ x,
                                const float* __restrict__ sc,
                                const float* __restrict__ sh,
                                unsigned short* __restrict__ y, size_t n) {
  size_t e = (size_t)blockIdx.x * 256 + threadIdx.x;
  if (e >= n) return;
  int c = (int)(e & 63);
  y[e] = f2bf(x[e] * sc[c] + sh[c]);
}

__global__ void k_bn_apply_f32(const float* __restrict__ x,
                               const float* __restrict__ sc,
                               const float* __restrict__ sh,
                               float* __restrict__ y, size_t n) {
  size_t e = (size_t)blockIdx.x * 256 + threadIdx.x;
  if (e >= n) return;
  int c = (int)(e & 63);
  y[e] = x[e] * sc[c] + sh[c];
}

// ---------------------------------------------------------------------------
// LayerNorm over last dim (64), bf16 out
// ---------------------------------------------------------------------------
__global__ void k_ln(const float* __restrict__ x, const float* __restrict__ g,
                     const float* __restrict__ b, unsigned short* __restrict__ y,
                     int M) {
  int m = blockIdx.x * blockDim.x + threadIdx.x;
  if (m >= M) return;
  const float* p = x + (size_t)m * 64;
  float s = 0.f, q = 0.f;
  for (int c = 0; c < 64; ++c) { float v = p[c]; s += v; q += v * v; }
  float mean = s * (1.f / 64.f);
  float var  = q * (1.f / 64.f) - mean * mean;
  float inv  = rsqrtf(var + 1e-3f);
  for (int c = 0; c < 64; ++c)
    y[(size_t)m * 64 + c] = f2bf((p[c] - mean) * inv * g[c] + b[c]);
}

// ---------------------------------------------------------------------------
// Flash attention over one axis.  Q/K/V: bf16 [B*C][256][256] (row = query /
// key index, col = feature).  Each block (128 thr = 4 waves) handles 64 query
// rows of one (b,c); wave owns 16 rows x 256 feature accumulator.
// V tile is staged LDS<-global by the Tensor Data Mover (TDM) when available.
// Output written as bf16 into [B*C][256][512] at column offset colOff.
// ---------------------------------------------------------------------------
__global__ __launch_bounds__(128)
void k_attn_flash(const unsigned short* __restrict__ qT,
                  const unsigned short* __restrict__ kT,
                  const unsigned short* __restrict__ vT,
                  unsigned short* __restrict__ outbf, int colOff) {
  __shared__ __align__(16) unsigned short Ks[256 * 32];  // [feat][j]   (K^T)
  __shared__ __align__(16) unsigned short Vs[32 * 256];  // [j][feat]
  __shared__ __align__(16) unsigned short Ps[4][16 * 32];// per-wave P pack
  const int tid = threadIdx.x, lane = tid & 31, wave = tid >> 5;
  const size_t bc = blockIdx.x;
  const int row0 = blockIdx.y * 64;
  const unsigned short* Qb = qT + bc * 65536;
  const unsigned short* Kb = kT + bc * 65536;
  const unsigned short* Vb = vT + bc * 65536;
  const int qrow = row0 + wave * 16 + (lane & 15);
  const unsigned short* qbase = Qb + (size_t)qrow * 256 + ((lane >> 4) << 3);

  const v8f vzero = {0.f, 0.f, 0.f, 0.f, 0.f, 0.f, 0.f, 0.f};
  v8f O[16];
#pragma unroll
  for (int ft = 0; ft < 16; ++ft) O[ft] = vzero;
  float mrow[8], lrow[8];
#pragma unroll
  for (int r = 0; r < 8; ++r) { mrow[r] = -3.0e38f; lrow[r] = 0.f; }
  const float scale = 0.0625f;   // 1/sqrt(256)

  for (int jt = 0; jt < 8; ++jt) {
    const int j0 = jt * 32;
    if (jt < 7) {  // prefetch next K tile (global_prefetch_b8)
      __builtin_prefetch(Kb + (size_t)(j0 + 32) * 256 + lane * 64, 0, 1);
    }
    // stage K^T ([feat][j]) by transposing gather
    for (int e = tid; e < 8192; e += 128) {
      int feat = e & 255, jj = e >> 8;
      Ks[feat * 32 + jj] = Kb[(size_t)(j0 + jj) * 256 + feat];
    }
    // stage V ([j][feat]): 32 x 256 bf16 tile -> TDM descriptor load
#if HAVE_TDM
    if (wave == 0) {
      unsigned lds_addr = (unsigned)(uintptr_t)(&Vs[0]);
      unsigned long long ga = (unsigned long long)(uintptr_t)(Vb + (size_t)j0 * 256);
      u32x4 g0;
      g0[0] = 1u;                                            // count=1 (valid D#)
      g0[1] = lds_addr;                                      // lds_addr
      g0[2] = (unsigned)(ga & 0xFFFFFFFFu);                  // global_addr[31:0]
      g0[3] = (unsigned)((ga >> 32) & 0x01FFFFFFu) | (2u << 30); // addr[56:32]|type=2
      i32x8 g1;
      g1[0] = 0x00010000;                                    // data_size = 2 bytes
      g1[1] = (int)(256u << 16);                             // tensor_dim0 = 256
      g1[2] = (int)(256u << 16);                             // tensor_dim1 = 256
      g1[3] = (int)(256u << 16);                             // tile_dim0 = 256
      g1[4] = 32;                                            // tile_dim1 = 32
      g1[5] = 256;                                           // tensor_dim0_stride
      g1[6] = 0; g1[7] = 0;
      i32x4 z4 = {0, 0, 0, 0};
      i32x8 z8 = {0, 0, 0, 0, 0, 0, 0, 0};
      __builtin_amdgcn_tensor_load_to_lds(g0, g1, z4, z4, z8, 0);
      __builtin_amdgcn_s_wait_tensorcnt(0);
    }
#else
    for (int e = tid; e < 8192; e += 128)
      Vs[e] = Vb[(size_t)j0 * 256 + e];
#endif
    __syncthreads();

    // S = Q * K^T for 32 kv columns (two 16x16 accumulators)
    v8f s0 = vzero, s1 = vzero;
#pragma unroll
    for (int ks = 0; ks < 8; ++ks) {
      FragBits fa;
      const unsigned short* qp = qbase + ks * 32;
      fa.q.lo = *(const uint4*)qp;
      fa.q.hi = *(const uint4*)(qp + 16);
      v16bf b0 = frag_b(Ks + (size_t)(ks * 32) * 32, 32);       // cols j0..j0+15
      v16bf b1 = frag_b(Ks + (size_t)(ks * 32) * 32 + 16, 32);  // cols +16..+31
      s0 = wmma_bf16(fa.v, b0, s0);
      s1 = wmma_bf16(fa.v, b1, s1);
    }

    // online softmax across the 32 new columns (rows live on 16-lane halves)
    float p0[8], p1[8], ratio[8];
#pragma unroll
    for (int r = 0; r < 8; ++r) {
      float cmax = fmaxf(s0[r], s1[r]) * scale;
#pragma unroll
      for (int off = 1; off < 16; off <<= 1)
        cmax = fmaxf(cmax, __shfl_xor(cmax, off, 32));
      float mn = fmaxf(mrow[r], cmax);
      ratio[r] = __expf(mrow[r] - mn);
      mrow[r] = mn;
      p0[r] = __expf(s0[r] * scale - mn);
      p1[r] = __expf(s1[r] * scale - mn);
      float sum = p0[r] + p1[r];
#pragma unroll
      for (int off = 1; off < 16; off <<= 1)
        sum += __shfl_xor(sum, off, 32);
      lrow[r] = lrow[r] * ratio[r] + sum;
    }
#pragma unroll
    for (int ft = 0; ft < 16; ++ft)
#pragma unroll
      for (int r = 0; r < 8; ++r) O[ft][r] *= ratio[r];

    // repack P (16x32) from C-layout to A-layout via per-wave LDS
    unsigned short* Pw = Ps[wave];
    {
      const int col = lane & 15, radd = (lane >> 4) << 3;
#pragma unroll
      for (int r = 0; r < 8; ++r) {
        int row = r + radd;
        Pw[row * 32 + col]      = f2bf(p0[r]);
        Pw[row * 32 + 16 + col] = f2bf(p1[r]);
      }
    }
    asm volatile("s_wait_dscnt 0" ::: "memory");   // same-wave LDS RAW
    v16bf ap = frag_a(Pw, 32);
#pragma unroll
    for (int ft = 0; ft < 16; ++ft) {
      v16bf bv = frag_b(Vs + ft * 16, 256);
      O[ft] = wmma_bf16(ap, bv, O[ft]);
    }
    __syncthreads();
  }

  const int col = lane & 15, radd = (lane >> 4) << 3;
  float inv[8];
#pragma unroll
  for (int r = 0; r < 8; ++r) inv[r] = 1.f / lrow[r];
#pragma unroll
  for (int ft = 0; ft < 16; ++ft) {
#pragma unroll
    for (int r = 0; r < 8; ++r) {
      int rg = row0 + wave * 16 + r + radd;
      outbf[(bc * 256 + rg) * 512 + colOff + ft * 16 + col] =
          f2bf(O[ft][r] * inv[r]);
    }
  }
}

// ---------------------------------------------------------------------------
// Orchestration
// ---------------------------------------------------------------------------
extern "C" void kernel_launch(void* const* d_in, const int* in_sizes, int n_in,
                              void* d_out, int out_size, void* d_ws, size_t ws_size,
                              hipStream_t stream) {
  (void)in_sizes; (void)n_in; (void)out_size; (void)ws_size;
  const float* inputs = (const float*)d_in[0];
  const float* wq = (const float*)d_in[1];  const float* bq = (const float*)d_in[2];
  const float* wk = (const float*)d_in[3];  const float* bk = (const float*)d_in[4];
  const float* wv = (const float*)d_in[5];  const float* bvv = (const float*)d_in[6];
  const float* gq = (const float*)d_in[7];  const float* betaq = (const float*)d_in[8];
  const float* gk = (const float*)d_in[9];  const float* betak = (const float*)d_in[10];
  const float* gv = (const float*)d_in[11]; const float* betav = (const float*)d_in[12];
  const float* wc = (const float*)d_in[13]; const float* bc = (const float*)d_in[14];
  const float* ln_g = (const float*)d_in[15]; const float* ln_b = (const float*)d_in[16];
  const float* w1 = (const float*)d_in[17]; const float* b1 = (const float*)d_in[18];
  const float* g1 = (const float*)d_in[19]; const float* beta1 = (const float*)d_in[20];
  const float* w2 = (const float*)d_in[21]; const float* b2 = (const float*)d_in[22];
  const float* g2 = (const float*)d_in[23]; const float* beta2 = (const float*)d_in[24];

  const size_t M1 = (size_t)8 * 256 * 256;   // 524288 pixels (stage 1 convs)
  const size_t M2 = (size_t)8 * 64 * 256;    // 131072 pixels (stage 2 convs)

  char* p = (char*)d_ws;
  auto alloc = [&](size_t bytes) -> void* {
    void* r = (void*)p; p += (bytes + 255) & ~(size_t)255; return r;
  };
  unsigned short* inp_bf = (unsigned short*)alloc(M1 * 64 * 2);
  unsigned short* wq_bf = (unsigned short*)alloc(64 * 576 * 2);
  unsigned short* wk_bf = (unsigned short*)alloc(64 * 576 * 2);
  unsigned short* wv_bf = (unsigned short*)alloc(64 * 576 * 2);
  unsigned short* wc_bf = (unsigned short*)alloc(64 * 4608 * 2);
  unsigned short* w1_bf = (unsigned short*)alloc(64 * 576 * 2);
  unsigned short* w2_bf = (unsigned short*)alloc(64 * 576 * 2);
  float* stat = (float*)alloc(256 * sizeof(float));
  float* stat_sum = stat, *stat_sq = stat + 64, *bn_scale = stat + 128, *bn_shift = stat + 192;
  float* conv_tmp = (float*)alloc(M1 * 64 * 4);        // reused by every conv
  unsigned short* qT1 = (unsigned short*)alloc(M1 * 64 * 2);
  unsigned short* qT2 = (unsigned short*)alloc(M1 * 64 * 2);
  unsigned short* kT1 = (unsigned short*)alloc(M1 * 64 * 2);
  unsigned short* kT2 = (unsigned short*)alloc(M1 * 64 * 2);
  unsigned short* vT1 = (unsigned short*)alloc(M1 * 64 * 2);
  unsigned short* vT2 = (unsigned short*)alloc(M1 * 64 * 2);
  unsigned short* attn_bf = (unsigned short*)alloc(M2 * 512 * 2); // [B,C,S,2S]
  unsigned short* ln_bf = (unsigned short*)alloc(M2 * 64 * 2);
  unsigned short* f1_bf = (unsigned short*)alloc(M2 * 64 * 2);

  // 1) inputs -> bf16; weights -> [Co][9*Ci] bf16
  {
    size_t n = M1 * 64;
    k_cvt_bf16<<<dim3((unsigned)((n + 255) / 256)), dim3(256), 0, stream>>>(inputs, inp_bf, n);
  }
  k_prep_w<<<dim3((9 * 64 * 64 + 255) / 256), dim3(256), 0, stream>>>(wq, wq_bf, 64, 64);
  k_prep_w<<<dim3((9 * 64 * 64 + 255) / 256), dim3(256), 0, stream>>>(wk, wk_bf, 64, 64);
  k_prep_w<<<dim3((9 * 64 * 64 + 255) / 256), dim3(256), 0, stream>>>(wv, wv_bf, 64, 64);
  k_prep_w<<<dim3((9 * 512 * 64 + 255) / 256), dim3(256), 0, stream>>>(wc, wc_bf, 512, 64);
  k_prep_w<<<dim3((9 * 64 * 64 + 255) / 256), dim3(256), 0, stream>>>(w1, w1_bf, 64, 64);
  k_prep_w<<<dim3((9 * 64 * 64 + 255) / 256), dim3(256), 0, stream>>>(w2, w2_bf, 64, 64);

  // 2) q/k/v conv + BN + dual-layout bf16 transpose
  unsigned short* wbfs[3] = {wq_bf, wk_bf, wv_bf};
  const float* biases[3] = {bq, bk, bvv};
  const float* gammas[3] = {gq, gk, gv};
  const float* betas[3]  = {betaq, betak, betav};
  unsigned short* T1s[3] = {qT1, kT1, vT1};
  unsigned short* T2s[3] = {qT2, kT2, vT2};
  for (int t = 0; t < 3; ++t) {
    k_conv3x3_wmma<<<dim3((unsigned)(M1 / 64)), dim3(128), 0, stream>>>(
        inp_bf, wbfs[t], biases[t], conv_tmp, 256, 256, 64, 0);
    (void)hipMemsetAsync(stat_sum, 0, 2 * 64 * sizeof(float), stream);
    k_bn_stats<<<dim3(1024), dim3(256), 0, stream>>>(conv_tmp, (int)M1, stat_sum, stat_sq);
    k_bn_finalize<<<dim3(1), dim3(64), 0, stream>>>(stat_sum, stat_sq, gammas[t], betas[t],
                                                    (float)M1, bn_scale, bn_shift);
    k_bn_apply_tr<<<dim3((unsigned)(M1 * 64 / 256)), dim3(256), 0, stream>>>(
        conv_tmp, bn_scale, bn_shift, T1s[t], T2s[t]);
  }

  // 3) axial attention (time then frequency), concat into [B,C,S,2S]
  k_attn_flash<<<dim3(512, 4), dim3(128), 0, stream>>>(qT1, kT1, vT1, attn_bf, 0);
  k_attn_flash<<<dim3(512, 4), dim3(128), 0, stream>>>(qT2, kT2, vT2, attn_bf, 256);

  // 4) conv wc (NHWC: N=8,H=64,W=256,Cin=512) + LayerNorm
  k_conv3x3_wmma<<<dim3((unsigned)(M2 / 64)), dim3(128), 0, stream>>>(
      attn_bf, wc_bf, bc, conv_tmp, 64, 256, 512, 0);
  k_ln<<<dim3((unsigned)((M2 + 255) / 256)), dim3(256), 0, stream>>>(
      conv_tmp, ln_g, ln_b, ln_bf, (int)M2);

  // 5) f1 = BN(relu(conv w1))
  k_conv3x3_wmma<<<dim3((unsigned)(M2 / 64)), dim3(128), 0, stream>>>(
      ln_bf, w1_bf, b1, conv_tmp, 64, 256, 64, 1);
  (void)hipMemsetAsync(stat_sum, 0, 2 * 64 * sizeof(float), stream);
  k_bn_stats<<<dim3(1024), dim3(256), 0, stream>>>(conv_tmp, (int)M2, stat_sum, stat_sq);
  k_bn_finalize<<<dim3(1), dim3(64), 0, stream>>>(stat_sum, stat_sq, g1, beta1,
                                                  (float)M2, bn_scale, bn_shift);
  k_bn_apply_bf16<<<dim3((unsigned)(M2 * 64 / 256)), dim3(256), 0, stream>>>(
      conv_tmp, bn_scale, bn_shift, f1_bf, M2 * 64);

  // 6) f2 = BN(relu(conv w2)) -> d_out (f32)
  k_conv3x3_wmma<<<dim3((unsigned)(M2 / 64)), dim3(128), 0, stream>>>(
      f1_bf, w2_bf, b2, conv_tmp, 64, 256, 64, 1);
  (void)hipMemsetAsync(stat_sum, 0, 2 * 64 * sizeof(float), stream);
  k_bn_stats<<<dim3(1024), dim3(256), 0, stream>>>(conv_tmp, (int)M2, stat_sum, stat_sq);
  k_bn_finalize<<<dim3(1), dim3(64), 0, stream>>>(stat_sum, stat_sq, g2, beta2,
                                                  (float)M2, bn_scale, bn_shift);
  k_bn_apply_f32<<<dim3((unsigned)(M2 * 64 / 256)), dim3(256), 0, stream>>>(
      conv_tmp, bn_scale, bn_shift, (float*)d_out, M2 * 64);
}
